// ScaledDotProductAttention_8847632630254
// MI455X (gfx1250) — matrix operand order repaired and checked by reading, not compiled, not run on hardware
//
#include <hip/hip_runtime.h>

#define B_N    4
#define S_N    4096
#define D_N    1024
#define MTILE  32
#define NK     64
#define BLOCK  256
#define SCALE  0.03125f   // 1/sqrt(1024)

typedef __attribute__((ext_vector_type(16))) __bf16 bf16x16;
typedef __attribute__((ext_vector_type(8)))  __bf16 bf16x8;
typedef __attribute__((ext_vector_type(8)))  float  f32x8;
typedef __attribute__((ext_vector_type(4)))  float  f32x4;

// ---- LDS layout (bytes). Padded strides avoid 64-bank conflicts. ----
#define Q_STRIDE   1032   // bf16 elems per Q row   (1024 + 8)
#define VT_STRIDE  72     // bf16 elems per V^T row (64 + 8), 144B = 16B aligned
#define PT_STRIDE  72
#define LDS_Q      0           // 32  * Q_STRIDE  * 2 = 66048
#define LDS_VT     66048       // 1024* VT_STRIDE * 2 = 147456
#define LDS_PT     213504      // 32  * PT_STRIDE * 2 = 4608
#define LDS_MAX    218112      // 2*4*16*4 = 512
#define LDS_SUM    218624      // 512
#define LDS_BYTES  219136      // ~214 KB of the 320 KB WGP LDS

static __device__ __forceinline__ bf16x16 join8(bf16x8 lo, bf16x8 hi) {
  return __builtin_shufflevector(lo, hi, 0,1,2,3,4,5,6,7,8,9,10,11,12,13,14,15);
}

static __device__ __forceinline__ bf16x16 cvt16(f32x4 a, f32x4 b, f32x4 c, f32x4 d) {
  bf16x16 r;
  r[0]=(__bf16)a.x;  r[1]=(__bf16)a.y;  r[2]=(__bf16)a.z;  r[3]=(__bf16)a.w;
  r[4]=(__bf16)b.x;  r[5]=(__bf16)b.y;  r[6]=(__bf16)b.z;  r[7]=(__bf16)b.w;
  r[8]=(__bf16)c.x;  r[9]=(__bf16)c.y;  r[10]=(__bf16)c.z; r[11]=(__bf16)c.w;
  r[12]=(__bf16)d.x; r[13]=(__bf16)d.y; r[14]=(__bf16)d.z; r[15]=(__bf16)d.w;
  return r;
}

// reductions within the two 16-lane halves of a wave32 (C-layout rows)
static __device__ __forceinline__ float red_max16(float v) {
  v = fmaxf(v, __shfl_xor(v, 1, 32));
  v = fmaxf(v, __shfl_xor(v, 2, 32));
  v = fmaxf(v, __shfl_xor(v, 4, 32));
  v = fmaxf(v, __shfl_xor(v, 8, 32));
  return v;
}
static __device__ __forceinline__ float red_sum16(float v) {
  v += __shfl_xor(v, 1, 32);
  v += __shfl_xor(v, 2, 32);
  v += __shfl_xor(v, 4, 32);
  v += __shfl_xor(v, 8, 32);
  return v;
}

__global__ __launch_bounds__(BLOCK)
void fattn_bf16_wmma(const float* __restrict__ q, const float* __restrict__ k,
                     const float* __restrict__ v, float* __restrict__ o) {
  extern __shared__ char smem[];
  __bf16* qT   = (__bf16*)(smem + LDS_Q);    // [32][Q_STRIDE]
  __bf16* vT   = (__bf16*)(smem + LDS_VT);   // [1024][VT_STRIDE]  (V transposed)
  __bf16* pT   = (__bf16*)(smem + LDS_PT);   // [32][PT_STRIDE]
  float*  rMax = (float*)(smem + LDS_MAX);   // [2][4][16]
  float*  rSum = (float*)(smem + LDS_SUM);   // [2][4][16]

  const int tid   = threadIdx.x;
  const int wave  = tid >> 5;
  const int lane  = tid & 31;
  const int half  = lane >> 4;     // which 16-lane half
  const int l16   = lane & 15;
  const int strip = wave >> 2;     // 0..1 : 16-row query strip
  const int kt    = wave & 3;      // 0..3 : QK^T key subtile == P@V d-slice
  const int b     = blockIdx.y;
  const int q0    = blockIdx.x * MTILE;

  const float* Qg = q + ((size_t)b * S_N + q0) * D_N;
  const float* Kg = k + (size_t)b * S_N * D_N;
  const float* Vg = v + (size_t)b * S_N * D_N;
  float*       Og = o + ((size_t)b * S_N + q0) * D_N;

  // ---- Stage Q (pre-scaled) into LDS as bf16, row-major ----
  for (int i = tid; i < MTILE * (D_N / 4); i += BLOCK) {
    int row = i >> 8;            // 256 float4 per row
    int c4  = i & 255;
    f32x4 val = *(const f32x4*)(Qg + (size_t)row * D_N + c4 * 4);
    __bf16* dst = qT + row * Q_STRIDE + c4 * 4;
    dst[0] = (__bf16)(val.x * SCALE);
    dst[1] = (__bf16)(val.y * SCALE);
    dst[2] = (__bf16)(val.z * SCALE);
    dst[3] = (__bf16)(val.w * SCALE);
  }

  f32x8 acc[16];                  // 16 d-tiles of 16 cols: this wave's O slice
#pragma unroll
  for (int t = 0; t < 16; ++t) { f32x8 z = {}; acc[t] = z; }
  float m_run[8], l_run[8];
#pragma unroll
  for (int r = 0; r < 8; ++r) { m_run[r] = -3.0e38f; l_run[r] = 0.0f; }

  const int arow_off = (strip * 16 + l16) * Q_STRIDE;

  for (int kb = 0; kb < S_N; kb += NK) {
    __syncthreads();   // prev iteration done reading vT/pT

    // ---- Stage V^T tile (fp32 -> bf16, transposed) ----
    for (int i = tid; i < NK * (D_N / 4); i += BLOCK) {
      int key = i >> 8;
      int c4  = i & 255;
      const float* src = Vg + (size_t)(kb + key) * D_N + c4 * 4;
      f32x4 val = *(const f32x4*)src;
      int d = c4 * 4;
      vT[(d + 0) * VT_STRIDE + key] = (__bf16)val.x;
      vT[(d + 1) * VT_STRIDE + key] = (__bf16)val.y;
      vT[(d + 2) * VT_STRIDE + key] = (__bf16)val.z;
      vT[(d + 3) * VT_STRIDE + key] = (__bf16)val.w;
      if (kb + NK < S_N) __builtin_prefetch(src + (size_t)NK * D_N, 0, 1);
    }

    // ---- QK^T: one 16x16 logit tile per wave, K-reduction over D=1024 ----
    f32x8 c8 = {};
    const float* kbase = Kg + (size_t)(kb + kt * 16 + l16) * D_N + half * 16;
#pragma unroll 4
    for (int dk = 0; dk < D_N; dk += 32) {
      // A (Q): lane = row, 16 contiguous K-elements at dk+half*16
      const __bf16* ap = qT + arow_off + dk + half * 16;
      bf16x8 alo = *(const bf16x8*)ap;
      bf16x8 ahi = *(const bf16x8*)(ap + 8);
      bf16x16 afrag = join8(alo, ahi);
      // B (K^T): lane = key column, 16 contiguous K-elements at dk+half*16
      const f32x4* kp = (const f32x4*)(kbase + dk);
      bf16x16 bfrag = cvt16(kp[0], kp[1], kp[2], kp[3]);
      c8 = __builtin_amdgcn_wmma_f32_16x16x32_bf16(false, afrag, false, bfrag,
                                                   (short)0, c8, false, false);
    }

    // ---- online softmax: per-tile row max ----
    float mloc[8];
#pragma unroll
    for (int r = 0; r < 8; ++r) mloc[r] = red_max16(c8[r]);
    if (l16 == 0) {
#pragma unroll
      for (int r = 0; r < 8; ++r)
        rMax[(strip * 4 + kt) * 16 + r + 8 * half] = mloc[r];
    }
    __syncthreads();

    float alpha[8];
#pragma unroll
    for (int r = 0; r < 8; ++r) {
      int row = r + 8 * half;
      float mt =        rMax[(strip * 4 + 0) * 16 + row];
      mt = fmaxf(mt,    rMax[(strip * 4 + 1) * 16 + row]);
      mt = fmaxf(mt,    rMax[(strip * 4 + 2) * 16 + row]);
      mt = fmaxf(mt,    rMax[(strip * 4 + 3) * 16 + row]);
      float mnew = fmaxf(m_run[r], mt);
      alpha[r] = __expf(m_run[r] - mnew);
      m_run[r] = mnew;
      c8[r] = __expf(c8[r] - mnew);
    }

    // ---- partial row sums + stage P (bf16) ----
#pragma unroll
    for (int r = 0; r < 8; ++r) {
      float s = red_sum16(c8[r]);
      if (l16 == 0) rSum[(strip * 4 + kt) * 16 + r + 8 * half] = s;
      pT[(strip * 16 + r + 8 * half) * PT_STRIDE + kt * 16 + l16] = (__bf16)c8[r];
    }
    __syncthreads();   // pT, rSum, vT all visible

    // ---- update l, rescale O ----
#pragma unroll
    for (int r = 0; r < 8; ++r) {
      int row = r + 8 * half;
      float s = rSum[(strip * 4 + 0) * 16 + row] + rSum[(strip * 4 + 1) * 16 + row]
              + rSum[(strip * 4 + 2) * 16 + row] + rSum[(strip * 4 + 3) * 16 + row];
      l_run[r] = l_run[r] * alpha[r] + s;
    }
#pragma unroll
    for (int t = 0; t < 16; ++t) {
#pragma unroll
      for (int r = 0; r < 8; ++r) acc[t][r] *= alpha[r];
    }

    // ---- P@V: this wave's strip x 256-col D slice ----
    const __bf16* prow = pT + (strip * 16 + l16) * PT_STRIDE;
    bf16x16 pa[2];
#pragma unroll
    for (int kk = 0; kk < 2; ++kk) {
      const __bf16* pp = prow + kk * 32 + half * 16;
      bf16x8 plo = *(const bf16x8*)pp;
      bf16x8 phi = *(const bf16x8*)(pp + 8);
      pa[kk] = join8(plo, phi);
    }
    const int dbase = kt * 256 + l16;
#pragma unroll
    for (int t = 0; t < 16; ++t) {
#pragma unroll
      for (int kk = 0; kk < 2; ++kk) {
        // B (V): lane = d column, 16 contiguous keys at kk*32+half*16 (32B LDS read)
        const __bf16* vp = vT + (dbase + t * 16) * VT_STRIDE + kk * 32 + half * 16;
        bf16x16 bfrag = *(const bf16x16*)vp;
        acc[t] = __builtin_amdgcn_wmma_f32_16x16x32_bf16(false, pa[kk], false, bfrag,
                                                         (short)0, acc[t], false, false);
      }
    }
  }

  // ---- epilogue: O = acc / l ----
#pragma unroll
  for (int r = 0; r < 8; ++r) {
    float inv = 1.0f / l_run[r];
    int row = strip * 16 + r + 8 * half;
    float* orow = Og + (size_t)row * D_N + kt * 256 + l16;
#pragma unroll
    for (int t = 0; t < 16; ++t) orow[t * 16] = acc[t][r] * inv;
  }
}

extern "C" void kernel_launch(void* const* d_in, const int* in_sizes, int n_in,
                              void* d_out, int out_size, void* d_ws, size_t ws_size,
                              hipStream_t stream) {
  const float* q = (const float*)d_in[0];
  const float* k = (const float*)d_in[1];
  const float* v = (const float*)d_in[2];
  float* o = (float*)d_out;
  dim3 grid(S_N / MTILE, B_N, 1);
  dim3 block(BLOCK, 1, 1);
  fattn_bf16_wmma<<<grid, block, LDS_BYTES, stream>>>(q, k, v, o);
}
